// MultiHeadLatentAttention_64381559767964
// MI455X (gfx1250) — compile-verified, matrix operand order checked by reading, not run on hardware
//
#include <hip/hip_runtime.h>
#include <hip/hip_bf16.h>

// ---------------------------------------------------------------------------
// MLA (Multi-head Latent Attention) for MI455X / gfx1250, wave32 + WMMA bf16.
//   q_c  = rmsnorm(x @ W_dq)                  [8192,1536]
//   kv_c = rmsnorm(x @ W_dkv)                 [8192, 512]
//   q    = concat(q_c@W_uq, rope(q_c@W_qr))   [8192,16,192]  (fused col-remap)
//   k    = concat(kv_c@W_uk, rope(x@W_kr))    [8192,16,192]
//   v    = kv_c@W_uv                          [8192,16,128]
//   attn = causal flash-attention (WMMA)      [8192,2048]
//   out  = attn @ W_o                         [8192,2048]
// All matmuls use v_wmma_f32_16x16x32_bf16 (f32 accum); f32 inputs are packed
// to bf16 with v_cvt_pk_bf16_f32 while staging tiles into LDS.
// ---------------------------------------------------------------------------

typedef __attribute__((ext_vector_type(16))) __bf16 v16bf;
typedef __attribute__((ext_vector_type(8)))  float  v8f;
typedef __attribute__((ext_vector_type(4)))  unsigned int u32x4;
typedef __attribute__((ext_vector_type(2)))  unsigned int u32x2;

// Fixed problem dimensions (compile-time so address math folds to immediates)
#define CT_T   2048
#define CT_H   16
#define CT_DH  128
#define CT_DR  64
#define CT_DQK 192

union Frag16 { v16bf v; u32x4 q[2]; };

// 16x32 bf16 WMMA operand fragment from an LDS tile row.
// kh = 8*(lane>>4). ISA 7.12.2: u[0..3] = K kh..kh+7, u[4..7] = K kh+16..kh+23.
__device__ inline v16bf load_frag(const __bf16* rowbase, int kh) {
    Frag16 f;
    f.q[0] = *(const u32x4*)(rowbase + kh);
    f.q[1] = *(const u32x4*)(rowbase + kh + 16);
    return f.v;
}

__device__ inline v8f zero8() {
    v8f z;
    #pragma unroll
    for (int i = 0; i < 8; ++i) z[i] = 0.0f;
    return z;
}

__device__ inline unsigned pack_bf16(float a, float b) {   // -> v_cvt_pk_bf16_f32
    union { __bf16 h[2]; unsigned u; } p;
    p.h[0] = (__bf16)a; p.h[1] = (__bf16)b;
    return p.u;
}

// ---------------------------------------------------------------------------
// Generic GEMM: C[M,N] = A[M,K] * B[K,N], f32 in/out, bf16 WMMA inside.
// GROUP==0: C[row*ldo + col].  GROUP>0: C[row*ldo + (col/GROUP)*gstride +
// goff + col%GROUP] (compile-time shift/and).  FULLN: N % 128 == 0.
// Tiles 128x128x32; 256 threads = 8 waves as 4(M) x 2(N); wave tile 32x64.
// ---------------------------------------------------------------------------
#define BLK_M 128
#define BLK_N 128
#define BLK_K 32

template<int GROUP, bool FULLN>
__global__ __launch_bounds__(256) void gemm_bf16_kernel(
    const float* __restrict__ A, const float* __restrict__ Bw,
    float* __restrict__ C, int M, int N, int K,
    int ldo, int gstride, int goff)
{
    __shared__ __align__(16) __bf16 As[BLK_M * BLK_K];
    __shared__ __align__(16) __bf16 Bt[BLK_N * BLK_K];   // B transposed: [n][k]

    const int tid  = threadIdx.x;
    const int lane = tid & 31;
    const int wave = tid >> 5;
    const int hi   = lane >> 4;
    const int lr   = lane & 15;
    const int kh   = hi * 8;

    const int bm = blockIdx.y * BLK_M;
    const int bn = blockIdx.x * BLK_N;
    const int wm = (wave >> 1) * 32;
    const int wn = (wave & 1) * 64;

    v8f acc[2][4];
    #pragma unroll
    for (int mi = 0; mi < 2; ++mi)
        #pragma unroll
        for (int nj = 0; nj < 4; ++nj) acc[mi][nj] = zero8();

    // A staging: thread -> 16 consecutive f32 of one row (4x float4)
    const int arow = tid >> 1, acb = (tid & 1) * 16;
    // B staging: thread -> one n-column, 16 k's (coalesced across lanes)
    const int bcol = tid & 127, bkb = (tid >> 7) * 16;
    const bool bok = FULLN || (bn + bcol < N);

    for (int k0 = 0; k0 < K; k0 += BLK_K) {
        {   // ---- A tile ----
            const float4* ap = (const float4*)(A + (size_t)(bm + arow) * K + k0 + acb);
            unsigned pk[8];
            #pragma unroll
            for (int j = 0; j < 4; ++j) {
                float4 f = ap[j];
                pk[2 * j]     = pack_bf16(f.x, f.y);
                pk[2 * j + 1] = pack_bf16(f.z, f.w);
            }
            u32x4* dst = (u32x4*)&As[arow * BLK_K + acb];
            dst[0] = *(u32x4*)&pk[0];
            dst[1] = *(u32x4*)&pk[4];
            if (k0 + BLK_K < K) __builtin_prefetch((const float*)ap + BLK_K, 0, 1);
        }
        {   // ---- B tile (transposed into Bt[n][k]) ----
            u32x4* dst = (u32x4*)&Bt[bcol * BLK_K + bkb];
            if (bok) {
                const float* bp = Bw + (size_t)(k0 + bkb) * N + bn + bcol;
                unsigned pk[8];
                #pragma unroll
                for (int j = 0; j < 8; ++j)
                    pk[j] = pack_bf16(bp[(size_t)(2 * j) * N], bp[(size_t)(2 * j + 1) * N]);
                dst[0] = *(u32x4*)&pk[0];
                dst[1] = *(u32x4*)&pk[4];
                if (k0 + BLK_K < K) __builtin_prefetch(bp + (size_t)BLK_K * N, 0, 1);
            } else {
                u32x4 z = (u32x4)(0u);
                dst[0] = z; dst[1] = z;
            }
        }
        __syncthreads();

        v16bf af[2], bf[4];
        #pragma unroll
        for (int mi = 0; mi < 2; ++mi)
            af[mi] = load_frag(&As[(wm + mi * 16 + lr) * BLK_K], kh);
        #pragma unroll
        for (int nj = 0; nj < 4; ++nj)
            bf[nj] = load_frag(&Bt[(wn + nj * 16 + lr) * BLK_K], kh);

        #pragma unroll
        for (int mi = 0; mi < 2; ++mi)
            #pragma unroll
            for (int nj = 0; nj < 4; ++nj)
                acc[mi][nj] = __builtin_amdgcn_wmma_f32_16x16x32_bf16(
                    false, af[mi], false, bf[nj], (short)0, acc[mi][nj], false, false);
        __syncthreads();
    }

    // store; C/D fragment layout: M = i + 8*hi, N = lr
    #pragma unroll
    for (int mi = 0; mi < 2; ++mi) {
        #pragma unroll
        for (int nj = 0; nj < 4; ++nj) {
            #pragma unroll
            for (int i = 0; i < 8; ++i) {
                int row = bm + wm + mi * 16 + i + hi * 8;
                int col = bn + wn + nj * 16 + lr;
                if (FULLN || col < N) {
                    size_t idx;
                    if (GROUP == 0)
                        idx = (size_t)row * ldo + col;
                    else
                        idx = (size_t)row * ldo + (size_t)(col / GROUP) * gstride
                              + goff + (col % GROUP);
                    C[idx] = acc[mi][nj][i];
                }
            }
        }
    }
}

// ---------------------------------------------------------------------------
// RMSNorm in place: one block per row.
// ---------------------------------------------------------------------------
__global__ __launch_bounds__(256) void rmsnorm_kernel(
    float* __restrict__ X, const float* __restrict__ g, int Ncols)
{
    __shared__ float red[256];
    float* xr = X + (size_t)blockIdx.x * Ncols;
    float s = 0.0f;
    for (int c = threadIdx.x; c < Ncols; c += 256) { float v = xr[c]; s += v * v; }
    red[threadIdx.x] = s;
    __syncthreads();
    for (int off = 128; off > 0; off >>= 1) {
        if (threadIdx.x < off) red[threadIdx.x] += red[threadIdx.x + off];
        __syncthreads();
    }
    float rinv = rsqrtf(red[0] / (float)Ncols + 1e-6f);
    for (int c = threadIdx.x; c < Ncols; c += 256) xr[c] = xr[c] * rinv * g[c];
}

// ---------------------------------------------------------------------------
// RoPE: ropes q tail [.,.,h,128:192] in place; ropes shared k_rope and
// broadcasts it into k[.,.,h,128:192] for all heads. One block per (b,t).
// Dimensions are compile-time constants so all strides fold to immediates.
// ---------------------------------------------------------------------------
__global__ __launch_bounds__(128) void rope_kernel(
    float* __restrict__ q, const float* __restrict__ kraw, float* __restrict__ k,
    const float* __restrict__ freqs /* [T][32][2] */)
{
    const int bt = blockIdx.x;
    const int t  = bt & (CT_T - 1);
    const int i  = threadIdx.x & 31;
    const int hq = threadIdx.x >> 5;

    const float c = freqs[(t * 32 + i) * 2 + 0];
    const float s = freqs[(t * 32 + i) * 2 + 1];

    const float k1 = kraw[(size_t)bt * CT_DR + 2 * i];
    const float k2 = kraw[(size_t)bt * CT_DR + 2 * i + 1];
    const float ko1 = k1 * c - k2 * s;
    const float ko2 = k1 * s + k2 * c;

    #pragma unroll
    for (int h = hq; h < CT_H; h += 4) {
        size_t idx = ((size_t)bt * CT_H + h) * CT_DQK + CT_DH + 2 * i;
        k[idx] = ko1; k[idx + 1] = ko2;
        float q1 = q[idx], q2 = q[idx + 1];
        q[idx]     = q1 * c - q2 * s;
        q[idx + 1] = q1 * s + q2 * c;
    }
}

// ---------------------------------------------------------------------------
// Causal flash attention, d_qk=192, d_v=128, WMMA bf16.
// One block = one (b,h) x 64 q-rows. 8 waves: wave w owns M-frag w>>1 and
// (for S) N-frag w&1; for O it owns 4 of the 8 dv fragments.
// All strides compile-time so staging loads use immediate offsets.
// ---------------------------------------------------------------------------
__global__ __launch_bounds__(256) void mla_attn_kernel(
    const float* __restrict__ Q, const float* __restrict__ Kb,
    const float* __restrict__ V, float* __restrict__ O)
{
    __shared__ __align__(16) __bf16 Qs[64 * CT_DQK];
    __shared__ __align__(16) __bf16 Ks[32 * CT_DQK];
    __shared__ __align__(16) float  Ss[64 * 32];
    __shared__ __align__(16) __bf16 Ps[64 * 32];
    __shared__ __align__(16) __bf16 Vts[CT_DH * 32];    // [dv][k]
    __shared__ float mrow[64], lrow[64], arow[64];

    const int tid  = threadIdx.x;
    const int lane = tid & 31;
    const int wave = tid >> 5;
    const int hi   = lane >> 4;
    const int lr   = lane & 15;
    const int kh   = hi * 8;

    constexpr int nqb = CT_T / 64;                       // 32
    const int qb  = blockIdx.x & (nqb - 1);
    const int h   = (blockIdx.x / nqb) & (CT_H - 1);
    const int b   = blockIdx.x / (nqb * CT_H);
    const int q0  = qb * 64;

    const float scale = 0.07216878364870322f;   // 1/sqrt(192)

    {   // stage Q: 12x float4 -> packed bf16 b64 stores
        int r = tid >> 2, cb = (tid & 3) * 48;
        const float4* qp = (const float4*)
            (Q + ((size_t)(b * CT_T + q0 + r) * CT_H + h) * CT_DQK + cb);
        #pragma unroll
        for (int j = 0; j < 12; ++j) {
            float4 f = qp[j];
            u32x2 u; u.x = pack_bf16(f.x, f.y); u.y = pack_bf16(f.z, f.w);
            *(u32x2*)&Qs[r * CT_DQK + cb + 4 * j] = u;
        }
    }
    if (tid < 64) { mrow[tid] = -1e30f; lrow[tid] = 0.0f; }

    const int wmi = wave >> 1;
    const int wnS = wave & 1;

    v8f oacc[4];
    #pragma unroll
    for (int j = 0; j < 4; ++j) oacc[j] = zero8();

    __syncthreads();

    const int kbmax = (q0 + 63) / 32;           // causal: skip blocks past diag
    for (int kb = 0; kb <= kbmax; ++kb) {
        const int k0 = kb * 32;
        {   // stage K: 6x float4 -> b64 stores
            int r = tid >> 3, cb = (tid & 7) * 24;
            const float4* kp = (const float4*)
                (Kb + ((size_t)(b * CT_T + k0 + r) * CT_H + h) * CT_DQK + cb);
            #pragma unroll
            for (int j = 0; j < 6; ++j) {
                float4 f = kp[j];
                u32x2 u; u.x = pack_bf16(f.x, f.y); u.y = pack_bf16(f.z, f.w);
                *(u32x2*)&Ks[r * CT_DQK + cb + 4 * j] = u;
            }
        }
        {   // stage V transposed to Vts[dv][k] via column walk:
            // thread -> one dv column (lane-coalesced); k-stride is the
            // compile-time constant CT_H*CT_DH so loads use immediate offsets
            int dv = tid & 127, khalf = (tid >> 7) * 16;
            const float* vp = V + ((size_t)(b * CT_T + k0 + khalf) * CT_H + h) * CT_DH + dv;
            constexpr int vstride = CT_H * CT_DH;        // 2048 floats
            unsigned pk[8];
            #pragma unroll
            for (int j = 0; j < 8; ++j)
                pk[j] = pack_bf16(vp[(2 * j) * vstride], vp[(2 * j + 1) * vstride]);
            u32x4* dst = (u32x4*)&Vts[dv * 32 + khalf];
            dst[0] = *(u32x4*)&pk[0];
            dst[1] = *(u32x4*)&pk[4];
        }
        __syncthreads();

        // S = Q K^T : one 16x16 fragment per wave, 6 K-steps of 32
        v8f sacc = zero8();
        #pragma unroll
        for (int s = 0; s < 6; ++s) {
            v16bf af = load_frag(&Qs[(wmi * 16 + lr) * CT_DQK + s * 32], kh);
            v16bf bf = load_frag(&Ks[(wnS * 16 + lr) * CT_DQK + s * 32], kh);
            sacc = __builtin_amdgcn_wmma_f32_16x16x32_bf16(
                false, af, false, bf, (short)0, sacc, false, false);
        }
        #pragma unroll
        for (int i = 0; i < 8; ++i) {           // scale + causal mask -> LDS
            int rl = wmi * 16 + i + hi * 8;
            int cl = wnS * 16 + lr;
            float sv = sacc[i] * scale;
            if (k0 + cl > q0 + rl) sv = -1e30f;
            Ss[rl * 32 + cl] = sv;
        }
        __syncthreads();

        if (tid < 64) {                         // online softmax, row-per-thread
            float mo = mrow[tid], mx = mo;
            #pragma unroll
            for (int c2 = 0; c2 < 32; ++c2) mx = fmaxf(mx, Ss[tid * 32 + c2]);
            float al = __expf(mo - mx), sum = 0.0f;
            #pragma unroll
            for (int c2 = 0; c2 < 32; c2 += 2) {
                float p0 = __expf(Ss[tid * 32 + c2]     - mx);
                float p1 = __expf(Ss[tid * 32 + c2 + 1] - mx);
                *(unsigned*)&Ps[tid * 32 + c2] = pack_bf16(p0, p1);
                sum += p0 + p1;
            }
            mrow[tid] = mx;
            lrow[tid] = lrow[tid] * al + sum;
            arow[tid] = al;
        }
        __syncthreads();

        // O = O*alpha + P @ V
        v16bf pf = load_frag(&Ps[(wmi * 16 + lr) * 32], kh);
        #pragma unroll
        for (int j = 0; j < 4; ++j) {
            int nb = (wave & 1) * 64 + j * 16;
            #pragma unroll
            for (int i = 0; i < 8; ++i) {
                int rl = wmi * 16 + i + hi * 8;
                oacc[j][i] = oacc[j][i] * arow[rl];
            }
            v16bf vf = load_frag(&Vts[(nb + lr) * 32], kh);
            oacc[j] = __builtin_amdgcn_wmma_f32_16x16x32_bf16(
                false, pf, false, vf, (short)0, oacc[j], false, false);
        }
        __syncthreads();
    }

    #pragma unroll
    for (int j = 0; j < 4; ++j) {
        int nb = (wave & 1) * 64 + j * 16;
        #pragma unroll
        for (int i = 0; i < 8; ++i) {
            int rl = wmi * 16 + i + hi * 8;
            int dv = nb + lr;
            O[((size_t)(b * CT_T + q0 + rl) * CT_H + h) * CT_DH + dv] =
                oacc[j][i] / lrow[rl];
        }
    }
}

// ---------------------------------------------------------------------------
// Host launch
// ---------------------------------------------------------------------------
template<int GROUP, bool FULLN>
static inline void launch_gemm(const float* A, const float* Bw, float* C,
                               int M, int N, int K, int ldo,
                               int gstride, int goff, hipStream_t s)
{
    dim3 grid((N + BLK_N - 1) / BLK_N, M / BLK_M);
    gemm_bf16_kernel<GROUP, FULLN><<<grid, 256, 0, s>>>(A, Bw, C, M, N, K, ldo, gstride, goff);
}

extern "C" void kernel_launch(void* const* d_in, const int* in_sizes, int n_in,
                              void* d_out, int out_size, void* d_ws, size_t ws_size,
                              hipStream_t stream)
{
    constexpr int Bz = 4, T = CT_T, DIM = 2048, QC = 1536, KVC = 512;
    constexpr int H = CT_H, DH = CT_DH, DR = CT_DR;
    constexpr int M = Bz * T;                    // 8192 token rows

    const float* x     = (const float*)d_in[0];
    const float* freqs = (const float*)d_in[2];  // [T,32,2]
    const float* W_dq  = (const float*)d_in[4];
    const float* W_uq  = (const float*)d_in[5];
    const float* W_dkv = (const float*)d_in[6];
    const float* W_uk  = (const float*)d_in[7];
    const float* W_uv  = (const float*)d_in[8];
    const float* W_qr  = (const float*)d_in[9];
    const float* W_kr  = (const float*)d_in[10];
    const float* W_o   = (const float*)d_in[11];
    const float* g_q   = (const float*)d_in[12];
    const float* g_kv  = (const float*)d_in[13];
    float* out = (float*)d_out;

    float* ws = (float*)d_ws;
    float* q_c  = ws;                                   // [M, QC]
    float* kv_c = q_c  + (size_t)M * QC;                // [M, KVC]
    float* q    = kv_c + (size_t)M * KVC;               // [M, H, 192]
    float* k    = q    + (size_t)M * H * (DH + DR);     // [M, H, 192]
    float* v    = k    + (size_t)M * H * (DH + DR);     // [M, H, 128]
    float* kraw = v    + (size_t)M * H * DH;            // [M, DR]
    float* attn = kraw + (size_t)M * DR;                // [M, H*DH]

    // q_c = rmsnorm(x @ W_dq)
    launch_gemm<0, true>(x, W_dq, q_c, M, QC, DIM, QC, 0, 0, stream);
    rmsnorm_kernel<<<M, 256, 0, stream>>>(q_c, g_q, QC);
    // kv_c = rmsnorm(x @ W_dkv)
    launch_gemm<0, true>(x, W_dkv, kv_c, M, KVC, DIM, KVC, 0, 0, stream);
    rmsnorm_kernel<<<M, 256, 0, stream>>>(kv_c, g_kv, KVC);
    // q nope -> q[:, h, 0:128]; q rope raw -> q[:, h, 128:192]
    launch_gemm<128, true>(q_c, W_uq, q, M, H * DH, QC, H * 192, 192, 0, stream);
    launch_gemm<64,  true>(q_c, W_qr, q, M, H * DR, QC, H * 192, 192, DH, stream);
    // k nope -> k[:, h, 0:128]; v; shared k_rope raw
    launch_gemm<128, true>(kv_c, W_uk, k, M, H * DH, KVC, H * 192, 192, 0, stream);
    launch_gemm<0,   true>(kv_c, W_uv, v, M, H * DH, KVC, H * DH, 0, 0, stream);
    launch_gemm<0,  false>(x, W_kr, kraw, M, DR, DIM, DR, 0, 0, stream);
    // rope q tails in place + broadcast roped k_rope into all heads of k
    rope_kernel<<<M, 128, 0, stream>>>(q, kraw, k, freqs);
    // causal flash attention
    mla_attn_kernel<<<Bz * H * (T / 64), 256, 0, stream>>>(q, k, v, attn);
    // output projection
    launch_gemm<0, true>(attn, W_o, out, M, DIM, H * DH, DIM, 0, 0, stream);
}
